// ScaledDotProductAttention_30107720745040
// MI455X (gfx1250) — matrix operand order repaired and checked by reading, not compiled, not run on hardware
//
#include <hip/hip_runtime.h>
#include <hip/hip_bf16.h>

typedef __attribute__((ext_vector_type(16))) _Float16 v16h;
typedef __attribute__((ext_vector_type(8)))  float    v8f;
typedef __attribute__((ext_vector_type(4)))  float    f32x4;

// Problem constants (from reference setup_inputs)
#define SB   2            // batch
#define SH   16           // heads
#define SS   2048         // sequence
#define SD   64           // head dim
#define ROW_PAD 2052      // 2048 + 4 floats padding: row stride %64 banks == 4 -> conflict-free row access
#define NTHREADS 128      // 4 wave32
#define SMEM_FLOATS (16*ROW_PAD + 16*64 + 128 + 16 + 16)

__global__ __launch_bounds__(NTHREADS)
void sdpa_wmma_kernel(const float* __restrict__ q,
                      const float* __restrict__ k,
                      const float* __restrict__ v,
                      const float* __restrict__ mask,
                      float* __restrict__ ctx_out,
                      float* __restrict__ attn_out)
{
    const int bh   = blockIdx.x >> 7;     // 128 row-tiles per (b,h)
    const int qm0  = (blockIdx.x & 127) << 4;
    const int tid  = threadIdx.x;
    const int wave = tid >> 5;
    const int lane = tid & 31;
    const int lN   = lane & 15;           // column (B/C/D) or row M (A) selector
    const int lH   = lane >> 4;           // half-wave selector

    extern __shared__ float smem[];
    float* sc     = smem;                 // [16][ROW_PAD] masked scores -> exp values
    float* ctx    = smem + 16*ROW_PAD;    // [16][64] partial context
    float* part   = ctx  + 16*64;         // [16][8] reduction partials
    float* rowmx  = part + 128;           // [16]
    float* rowinv = rowmx + 16;           // [16] 1/rowsum

    for (int i = tid; i < 16*64; i += NTHREADS) ctx[i] = 0.0f;

    // ---- Load Q tile as two A-fragments (K = 0..31 and 32..63), fold 1/sqrt(64) ----
    // A layout: lane -> M = lane%16; element e -> K = 16*(lane/16) + e
    v16h qa0, qa1;
    {
        const float* qrow = q + ((size_t)bh * SS + qm0 + lN) * SD;
        #pragma unroll
        for (int e = 0; e < 16; ++e) {
            int j = e >> 1, p = e & 1;
            int kk = (lH << 4) + (j << 1) + p;
            qa0[e] = (_Float16)(qrow[kk]      * 0.125f);
            qa1[e] = (_Float16)(qrow[kk + 32] * 0.125f);
        }
    }

    // ================= Phase 1: scores = (Q K^T)/8 * mask -> LDS =================
    // Each wave owns 32 column tiles (512 columns).
    for (int ct = wave * 32; ct < wave * 32 + 32; ++ct) {
        const int n0 = ct << 4;
        // B layout (32x16): lane -> N = lane%16; element e -> K = 16*(lane/16) + e
        v16h kb0, kb1;
        const float* krow = k + ((size_t)bh * SS + n0 + lN) * SD + (lH << 4);
        #pragma unroll
        for (int e = 0; e < 16; ++e) {
            kb0[e] = (_Float16)krow[e];
            kb1[e] = (_Float16)krow[e + 32];
        }
        v8f acc = {};
        acc = __builtin_amdgcn_wmma_f32_16x16x32_f16(false, qa0, false, kb0, (short)0, acc, false, false);
        acc = __builtin_amdgcn_wmma_f32_16x16x32_f16(false, qa1, false, kb1, (short)0, acc, false, false);
        // C/D layout: vgpr r -> row M = r + 8*(lane/16), col N = lane%16
        // Mask is streamed use-once (536 MB): non-temporal loads keep K/V resident in L2.
        const float* mrow = mask + ((size_t)bh * SS + qm0 + (lH << 3)) * SS + n0 + lN;
        #pragma unroll
        for (int r = 0; r < 8; ++r) {
            float mv = __builtin_nontemporal_load(mrow + (size_t)r * SS);
            sc[(r + (lH << 3)) * ROW_PAD + n0 + lN] = acc[r] * mv;
        }
    }
    __syncthreads();

    // ================= Phase 2: exact softmax over LDS strip (b128 scans) ========
    {   // row max
        const int row = tid & 15, seg = tid >> 4;
        const f32x4* p = reinterpret_cast<const f32x4*>(sc + row * ROW_PAD + seg * 256);
        f32x4 m4 = p[0];
        for (int j = 1; j < 64; ++j) {
            m4 = __builtin_elementwise_max(m4, p[j]);
        }
        part[row * 8 + seg] = fmaxf(fmaxf(m4[0], m4[1]), fmaxf(m4[2], m4[3]));
    }
    __syncthreads();
    if (tid < 16) {
        float m = -3.4e38f;
        #pragma unroll
        for (int j = 0; j < 8; ++j) m = fmaxf(m, part[tid * 8 + j]);
        rowmx[tid] = m;
    }
    __syncthreads();
    {   // exp + row sum (in-place, b128)
        const int row = tid & 15, seg = tid >> 4;
        f32x4* p = reinterpret_cast<f32x4*>(sc + row * ROW_PAD + seg * 256);
        const float mx = rowmx[row];
        f32x4 s4 = {0.0f, 0.0f, 0.0f, 0.0f};
        for (int j = 0; j < 64; ++j) {
            f32x4 x = p[j];
            f32x4 e;
            e[0] = __expf(x[0] - mx);
            e[1] = __expf(x[1] - mx);
            e[2] = __expf(x[2] - mx);
            e[3] = __expf(x[3] - mx);
            p[j] = e;
            s4 += e;
        }
        part[row * 8 + seg] = (s4[0] + s4[1]) + (s4[2] + s4[3]);
    }
    __syncthreads();
    if (tid < 16) {
        float s = 0.0f;
        #pragma unroll
        for (int j = 0; j < 8; ++j) s += part[tid * 8 + j];
        rowinv[tid] = 1.0f / s;
    }
    __syncthreads();

    // ================= Phase 3a: stream normalized attn to HBM ==================
    // 536 MB use-once stream: non-temporal b128 stores, issued before the PV WMMAs
    // so STOREcnt drains while phase 3b computes.
    for (int idx = tid; idx < 16 * (SS / 4); idx += NTHREADS) {
        const int row = idx >> 9;
        const int c4  = (idx & 511) << 2;
        f32x4 x = *reinterpret_cast<const f32x4*>(sc + row * ROW_PAD + c4);
        f32x4 o = x * rowinv[row];
        __builtin_nontemporal_store(
            o, reinterpret_cast<f32x4*>(attn_out + ((size_t)bh * SS + qm0 + row) * SS + c4));
    }

    // ================= Phase 3b: context = exp(scores) @ V via WMMA =============
    v8f cacc[4] = {v8f{}, v8f{}, v8f{}, v8f{}};
    for (int chunk = 0; chunk < 16; ++chunk) {
        const int col0 = wave * 512 + chunk * 32;
        // A fragment from LDS exp values: M = lane%16 (query row), K = key column
        v16h a;
        const float* arow = sc + lN * ROW_PAD + col0 + (lH << 4);
        #pragma unroll
        for (int e = 0; e < 16; ++e) {
            int j = e >> 1, p = e & 1;
            int kk = (j << 1) + p;
            a[e] = (_Float16)arow[kk];
        }
        #pragma unroll
        for (int dt = 0; dt < 4; ++dt) {
            // B fragment from V: K = key row (col0 + 16*lH + e), N = dt*16 + lane%16
            v16h bf;
            const float* vrow = v + ((size_t)bh * SS + col0 + (lH << 4)) * SD + dt * 16 + lN;
            #pragma unroll
            for (int e = 0; e < 16; ++e) bf[e] = (_Float16)vrow[(size_t)e * SD];
            cacc[dt] = __builtin_amdgcn_wmma_f32_16x16x32_f16(false, a, false, bf, (short)0, cacc[dt], false, false);
        }
    }
    // cross-wave reduction of the 16x64 partial context via LDS float atomics
    #pragma unroll
    for (int dt = 0; dt < 4; ++dt) {
        #pragma unroll
        for (int r = 0; r < 8; ++r) {
            atomicAdd(&ctx[(r + (lH << 3)) * 64 + dt * 16 + lN], cacc[dt][r]);
        }
    }
    __syncthreads();

    for (int idx = tid; idx < 16 * 64; idx += NTHREADS) {
        const int row = idx >> 6;
        const int d   = idx & 63;
        ctx_out[((size_t)bh * SS + qm0 + row) * SD + d] = ctx[idx] * rowinv[row];
    }
}

extern "C" void kernel_launch(void* const* d_in, const int* in_sizes, int n_in,
                              void* d_out, int out_size, void* d_ws, size_t ws_size,
                              hipStream_t stream) {
    const float* q    = (const float*)d_in[0];
    const float* k    = (const float*)d_in[1];
    const float* v    = (const float*)d_in[2];
    const float* mask = (const float*)d_in[3];

    // Output tuple flattened in return order: context (B,H,S,D) then attn (B,H,S,S)
    float* ctx_out  = (float*)d_out;
    float* attn_out = (float*)d_out + (size_t)SB * SH * SS * SD;

    const int smem_bytes = SMEM_FLOATS * sizeof(float); // ~136 KB (LDS is 320 KB/WGP on CDNA5)
    hipFuncSetAttribute((const void*)sdpa_wmma_kernel,
                        hipFuncAttributeMaxDynamicSharedMemorySize, smem_bytes);

    dim3 grid(SB * SH * (SS / 16));   // 4096 workgroups
    dim3 block(NTHREADS);             // 4 wave32
    sdpa_wmma_kernel<<<grid, block, smem_bytes, stream>>>(q, k, v, mask, ctx_out, attn_out);
}
